// GATModel_29703993819800
// MI455X (gfx1250) — compile-verified, matrix operand order checked by reading, not run on hardware
//
#include <hip/hip_runtime.h>
#include <hip/hip_bf16.h>
#include <stdint.h>

// Problem constants (match reference)
#define Bg   4096          // graphs
#define Pn   64            // nodes per graph (A + OBS = 8 + 56)
#define Ntot (Bg * Pn)     // 262144 nodes
#define EPG  56            // edges per graph
#define Etot (Bg * EPG)    // 229376 edges
#define HC   256           // H * HID = 8*32
#define NHEAD 8
#define HID  32
#define MAXE (EPG + Pn)    // 120 with self loops

typedef __attribute__((ext_vector_type(16))) _Float16     v16h;
typedef __attribute__((ext_vector_type(8)))  float        v8f;
typedef __attribute__((ext_vector_type(4)))  unsigned int v4u;

// ---------------------------------------------------------------------------
// Pack weight matrix W (Kreal x 256, f32 row-major) into WMMA B-fragment
// order (f16), zero-padding K up to Kpad.  Per (ktile,ntile) 32x16 tile:
// 32 lanes x 16 halves contiguous; half i of lane l is
//   k = ktile*32 + (l>>4)*8 + (i&7) + (i>>3)*16 ,  n = ntile*16 + (l&15)
// ---------------------------------------------------------------------------
__global__ void pack_w_kernel(const float* __restrict__ W, _Float16* __restrict__ Wp,
                              int Kreal, int Kpad) {
  int total = (Kpad / 32) * 16 * 512;
  int i = blockIdx.x * 256 + threadIdx.x;
  if (i >= total) return;
  int tIdx = i >> 9;
  int r    = i & 511;
  int lane = r >> 4;
  int ii   = r & 15;
  int kt   = tIdx >> 4;
  int nt   = tIdx & 15;
  int k = kt * 32 + ((lane >> 4) * 8) + (ii & 7) + ((ii >> 3) * 16);
  int n = nt * 16 + (lane & 15);
  Wp[i] = (k < Kreal) ? (_Float16)W[k * HC + n] : (_Float16)0.0f;
}

// Convert x (N x 16 f32) -> x16 (N x 32 f16, K zero-padded to 32)
__global__ void conv_x_kernel(const float* __restrict__ x, _Float16* __restrict__ x16) {
  long i = (long)blockIdx.x * 256 + threadIdx.x;    // over N*32
  if (i >= (long)Ntot * 32) return;
  int  k = (int)(i & 31);
  long n = i >> 5;
  x16[i] = (k < 16) ? (_Float16)x[n * 16 + k] : (_Float16)0.0f;
}

// Per-node self-loop edge attr: ea_loop[n] = segsum(ea,dst)/max(cnt,1)
__global__ __launch_bounds__(64) void ealoop_kernel(const int* __restrict__ edst,
                                                    const float* __restrict__ ea,
                                                    float* __restrict__ eal) {
  __shared__ float sum[Pn];
  __shared__ int   cnt[Pn];
  int g = blockIdx.x, t = threadIdx.x;
  sum[t] = 0.0f; cnt[t] = 0;
  __syncthreads();
  if (t < EPG) {
    int d = edst[g * EPG + t] - g * Pn;
    atomicAdd(&cnt[d], 1);
    atomicAdd(&sum[d], ea[g * EPG + t]);
  }
  __syncthreads();
  eal[(long)g * Pn + t] = sum[t] / fmaxf((float)cnt[t], 1.0f);
}

// ---------------------------------------------------------------------------
// WMMA GEMM: C16(Nx256,f16) = A(NxK,f16) * Bpacked(Kx256,f16), f32 accumulate.
// 256 threads = 8 waves in 4(row) x 2(col) grid; wave = 16 rows x 128 cols =
// 8 v_wmma_f32_16x16x32_f16 tiles. Software pipelined (B ping-pong groups of
// 4 tiles, A double buffered). Epilogue transposes each wave's tile through
// padded LDS so the f16 output is written with fully-coalesced b128 stores.
// ---------------------------------------------------------------------------
__device__ __forceinline__ v16h load_frag(const _Float16* p, int secondOfs, bool nt) {
  union { v16h v; v4u q[2]; } u;
  if (nt) {
    u.q[0] = __builtin_nontemporal_load((const v4u*)p);
    u.q[1] = __builtin_nontemporal_load((const v4u*)(p + secondOfs));
  } else {
    u.q[0] = *(const v4u*)p;
    u.q[1] = *(const v4u*)(p + secondOfs);
  }
  return u.v;
}

#define STG_W 136   // padded row stride (halves): 272B rows -> <=2-way LDS bank conflicts

template <int K>
__global__ __launch_bounds__(256) void gemm_wmma_kernel(const _Float16* __restrict__ A,
                                                        const _Float16* __restrict__ Bp,
                                                        _Float16* __restrict__ C16) {
  __shared__ _Float16 stg[8][16][STG_W];   // ~34 KB, per-wave private regions
  int t = threadIdx.x;
  int wave = t >> 5, lane = t & 31;
  int rowg = wave & 3;
  int colg = wave >> 2;
  long rb = (long)blockIdx.x * 64 + rowg * 16;
  int  cb = colg * 128;
  int  lr = lane & 15;
  int  lh = lane >> 4;

  const _Float16* aBase = A + (rb + lr) * (long)K + lh * 8;
  const _Float16* bBase = Bp + ((long)(cb >> 4)) * 512 + lane * 16;

  constexpr int NSTEP = K / 32;
  v8f acc[8] = {};
  v16h aCur, aNxt;
  v16h b0[4], b1[4];

  aCur = load_frag(aBase, 16, true);                     // A streamed: NT
#pragma unroll
  for (int q = 0; q < 4; ++q)
    b0[q] = load_frag(bBase + (long)q * 512, 8, false);  // W reused: cached

#pragma unroll
  for (int s = 0; s < NSTEP; ++s) {
    const _Float16* bStep = bBase + (long)s * (16 * 512);
#pragma unroll
    for (int q = 0; q < 4; ++q)
      b1[q] = load_frag(bStep + (long)(4 + q) * 512, 8, false);
#pragma unroll
    for (int q = 0; q < 4; ++q)
      acc[q] = __builtin_amdgcn_wmma_f32_16x16x32_f16(
          false, aCur, false, b0[q], (short)0, acc[q], false, false);
    if (s + 1 < NSTEP) {
      aNxt = load_frag(aBase + (s + 1) * 32, 16, true);
      const _Float16* bNext = bBase + (long)(s + 1) * (16 * 512);
#pragma unroll
      for (int q = 0; q < 4; ++q)
        b0[q] = load_frag(bNext + (long)q * 512, 8, false);
    }
#pragma unroll
    for (int q = 0; q < 4; ++q)
      acc[4 + q] = __builtin_amdgcn_wmma_f32_16x16x32_f16(
          false, aCur, false, b1[q], (short)0, acc[4 + q], false, false);
    if (s + 1 < NSTEP) aCur = aNxt;
  }

  // Epilogue: transpose wave tile through LDS, then coalesced b128 f16 stores.
  // D layout: lane holds column lr, rows v + 8*lh.
  _Float16 (*my)[STG_W] = stg[wave];
#pragma unroll
  for (int tn = 0; tn < 8; ++tn)
#pragma unroll
    for (int v = 0; v < 8; ++v)
      my[v + 8 * lh][tn * 16 + lr] = (_Float16)acc[tn][v];
  // same-wave LDS ops are in order; no cross-wave sharing -> no barrier needed
  {
    int r  = lane >> 1;          // 0..15: row within wave tile
    int hf = lane & 1;           // half of the 128 columns
    const _Float16* sp = &my[r][hf * 64];
    _Float16* dp = C16 + (rb + r) * (long)HC + cb + hf * 64;
#pragma unroll
    for (int j = 0; j < 4; ++j)
      *(v4u*)(dp + j * 8) = *(const v4u*)(sp + j * 8);
  }
}

// ---------------------------------------------------------------------------
// GAT attention + aggregation, one workgroup per graph (64 nodes, <=120 edges)
// ---------------------------------------------------------------------------
__device__ __forceinline__ unsigned ordF(float f) {
  unsigned u = __float_as_uint(f);
  return (u & 0x80000000u) ? ~u : (u | 0x80000000u);
}
__device__ __forceinline__ float unordF(unsigned u) {
  return __uint_as_float((u & 0x80000000u) ? (u & 0x7fffffffu) : ~u);
}

__global__ __launch_bounds__(256) void gat_aggregate_kernel(
    const _Float16* __restrict__ hlin16,  // N x 256 (GEMM output, f16)
    const float* __restrict__ We,         // 256
    const float* __restrict__ a_s,        // 8x32
    const float* __restrict__ a_d,        // 8x32
    const float* __restrict__ a_e,        // 8x32
    const float* __restrict__ bias,       // 256
    const int*   __restrict__ esrc,       // E global ids
    const int*   __restrict__ edst,       // E global ids
    const float* __restrict__ ea,         // E
    const float* __restrict__ eal,        // N self-loop edge attr
    int selfLoops,
    _Float16* __restrict__ out16) {       // N x 256
  __shared__ _Float16 hls[Pn][HC];        // 32 KB staged features
  __shared__ float ss[Pn][NHEAD], sd[Pn][NHEAD];
  __shared__ float logi[MAXE][NHEAD];     // logits, then p
  __shared__ unsigned mU[Pn][NHEAD];      // ordered-float max
  __shared__ float denom[Pn][NHEAD];
  __shared__ float ce[NHEAD];
  __shared__ int   srcL[MAXE], dstL[MAXE];
  __shared__ float eaL[MAXE];
  __shared__ int   cnt[Pn], startO[Pn + 1], fill[Pn];
  __shared__ short lst[MAXE];

  int g = blockIdx.x, t = threadIdx.x;
  int nE = selfLoops ? MAXE : EPG;
  long gn = (long)g * Pn;
  long gb = gn * HC;

  // stage node features to LDS: straight b128 copy (already f16)
  {
    const v4u* s4 = (const v4u*)(hlin16 + gb);
    v4u* d4 = (v4u*)&hls[0][0];
    for (int i = t; i < Pn * HC / 8; i += 256) d4[i] = s4[i];
  }

  // edges (+ self loops)
  for (int e = t; e < nE; e += 256) {
    if (e < EPG) {
      srcL[e] = esrc[g * EPG + e] - g * Pn;
      dstL[e] = edst[g * EPG + e] - g * Pn;
      eaL[e]  = ea[g * EPG + e];
    } else {
      int n = e - EPG;
      srcL[e] = n; dstL[e] = n;
      eaL[e]  = eal[gn + n];
    }
  }
  if (t < NHEAD) {
    float s = 0.0f;
    for (int j = 0; j < HID; ++j) s += We[t * HID + j] * a_e[t * HID + j];
    ce[t] = s;
  }
  for (int i = t; i < Pn * NHEAD; i += 256) {
    (&mU[0][0])[i] = 0u;            // ordF of any real float > 0, so 0 == "empty"
    (&denom[0][0])[i] = 0.0f;
  }
  if (t < Pn) { cnt[t] = 0; fill[t] = 0; }
  __syncthreads();

  // per-(node,head) attention dots
  for (int i = t; i < Pn * NHEAD; i += 256) {
    int n = i >> 3, h = i & 7;
    float s1 = 0.0f, s2 = 0.0f;
    for (int j = 0; j < HID; ++j) {
      float hv = (float)hls[n][h * HID + j];
      s1 += hv * a_s[h * HID + j];
      s2 += hv * a_d[h * HID + j];
    }
    ss[n][h] = s1; sd[n][h] = s2;
  }
  __syncthreads();

  // pass 1: leaky-relu logits, running max, degree count
  for (int e = t; e < nE; e += 256) {
    int s = srcL[e], d0 = dstL[e];
    float eav = eaL[e];
    atomicAdd(&cnt[d0], 1);
#pragma unroll
    for (int h = 0; h < NHEAD; ++h) {
      float l = ss[s][h] + sd[d0][h] + eav * ce[h];
      l = (l > 0.0f) ? l : 0.2f * l;
      logi[e][h] = l;
      atomicMax(&mU[d0][h], ordF(l));
    }
  }
  __syncthreads();

  if (t == 0) {                      // tiny serial prefix over 64 nodes
    int acc = 0;
    for (int n = 0; n < Pn; ++n) { startO[n] = acc; acc += cnt[n]; }
    startO[Pn] = acc;
  }
  __syncthreads();

  // pass 2: p = exp(l - m), denominators, CSR scatter
  for (int e = t; e < nE; e += 256) {
    int d0 = dstL[e];
    int pos = atomicAdd(&fill[d0], 1);
    lst[startO[d0] + pos] = (short)e;
#pragma unroll
    for (int h = 0; h < NHEAD; ++h) {
      float m = unordF(mU[d0][h]);
      float p = __expf(logi[e][h] - m);
      logi[e][h] = p;
      atomicAdd(&denom[d0][h], p);
    }
  }
  __syncthreads();

  // pass 3: atomic-free gather; thread t owns node t>>2, heads {t&3, (t&3)+4}
  int n  = t >> 2, q = t & 3;
  int s0 = startO[n], s1 = startO[n + 1];
  for (int hh = 0; hh < 2; ++hh) {
    int h = q + hh * 4;
    float acc[HID];
#pragma unroll
    for (int j = 0; j < HID; ++j) acc[j] = 0.0f;
    float dn = denom[n][h] + 1e-16f;
    for (int idx = s0; idx < s1; ++idx) {
      int e = lst[idx];
      float alpha = logi[e][h] / dn;
      int sN = srcL[e];
#pragma unroll
      for (int j = 0; j < HID; ++j)
        acc[j] += alpha * (float)hls[sN][h * HID + j];
    }
    long ob = gb + (long)n * HC + h * HID;
#pragma unroll
    for (int j = 0; j < HID; ++j) {
      float v = acc[j] + bias[h * HID + j];
      v = (v > 0.0f) ? v : 0.0f;     // relu after every GAT layer
      out16[ob + j] = (_Float16)v;
    }
  }
}

// ---------------------------------------------------------------------------
// Head: per-graph mean pool + fc1 (512->32, relu) + fc2 (32->2), f16 input
// ---------------------------------------------------------------------------
__global__ __launch_bounds__(256) void gat_head_kernel(
    const _Float16* __restrict__ h3, const float* __restrict__ fc1w,
    const float* __restrict__ fc1b, const float* __restrict__ fc2w,
    const float* __restrict__ fc2b, float* __restrict__ out) {
  __shared__ float emb[HC];
  __shared__ float z[8][32];
  int g = blockIdx.x, t = threadIdx.x;
  long gb = (long)g * Pn * HC;

  float s = 0.0f;                       // mean over 64 nodes, channel t
  for (int nn = 0; nn < Pn; ++nn) s += (float)h3[gb + (long)nn * HC + t];
  emb[t] = s * (1.0f / (float)Pn);
  __syncthreads();

  int a = t >> 5, o = t & 31;           // 8 agents x 32 outputs
  float acc = fc1b[o];
  for (int c = 0; c < HC; ++c) acc += (float)h3[gb + (long)a * HC + c] * fc1w[c * 32 + o];
  for (int c = 0; c < HC; ++c) acc += emb[c] * fc1w[(HC + c) * 32 + o];
  z[a][o] = (acc > 0.0f) ? acc : 0.0f;
  __syncthreads();

  if (t < 16) {
    int aa = t >> 1, oo = t & 1;
    float r = fc2b[oo];
    for (int k = 0; k < 32; ++k) r += z[aa][k] * fc2w[k * 2 + oo];
    out[((long)g * 8 + aa) * 2 + oo] = r;
  }
}

// ---------------------------------------------------------------------------
extern "C" void kernel_launch(void* const* d_in, const int* in_sizes, int n_in,
                              void* d_out, int out_size, void* d_ws, size_t ws_size,
                              hipStream_t stream) {
  const float* x   = (const float*)d_in[0];
  const int*   ei  = (const int*)d_in[1];
  const float* ea  = (const float*)d_in[2];
  const int* esrc = ei;
  const int* edst = ei + Etot;

  const float *W[3], *We[3], *As[3], *Ad[3], *Ae[3], *Bb[3];
  for (int l = 0; l < 3; ++l) {
    int base = 3 + l * 6;
    W[l]  = (const float*)d_in[base + 0];
    We[l] = (const float*)d_in[base + 1];
    As[l] = (const float*)d_in[base + 2];
    Ad[l] = (const float*)d_in[base + 3];
    Ae[l] = (const float*)d_in[base + 4];
    Bb[l] = (const float*)d_in[base + 5];
  }
  const float* fc1w = (const float*)d_in[21];
  const float* fc1b = (const float*)d_in[22];
  const float* fc2w = (const float*)d_in[23];
  const float* fc2b = (const float*)d_in[24];
  float* out = (float*)d_out;

  // Workspace layout: two ping-pong f16 activation buffers + packed W + eal
  char* ws = (char*)d_ws;
  const size_t SZ_H16 = (size_t)Ntot * HC * sizeof(_Float16);   // 128 MB each
  const size_t SZ_WP  = 8 * 16 * 512 * sizeof(_Float16);        // 128 KB per layer
  _Float16* bufA = (_Float16*)ws;                 // GEMM outputs
  _Float16* bufB = (_Float16*)(ws + SZ_H16);      // aggregate outputs / GEMM inputs
  _Float16* wp[3];
  wp[0] = (_Float16*)(ws + 2 * SZ_H16);
  wp[1] = (_Float16*)(ws + 2 * SZ_H16 + SZ_WP);
  wp[2] = (_Float16*)(ws + 2 * SZ_H16 + 2 * SZ_WP);
  float* eal = (float*)(ws + 2 * SZ_H16 + 3 * SZ_WP);           // N f32

  // 1) pack weights (layer 1 zero-padded K 16->32)
  pack_w_kernel<<<(1 * 16 * 512 + 255) / 256, 256, 0, stream>>>(W[0], wp[0], 16, 32);
  pack_w_kernel<<<(8 * 16 * 512 + 255) / 256, 256, 0, stream>>>(W[1], wp[1], HC, HC);
  pack_w_kernel<<<(8 * 16 * 512 + 255) / 256, 256, 0, stream>>>(W[2], wp[2], HC, HC);

  // 2) x -> f16 padded (stored at start of bufB, consumed by layer-1 GEMM)
  conv_x_kernel<<<(Ntot * 32 + 255) / 256, 256, 0, stream>>>(x, bufB);

  // 3) self-loop edge attrs (used by layers 2/3)
  ealoop_kernel<<<Bg, 64, 0, stream>>>(edst, ea, eal);

  // ---- layer 1 (no self loops) ----
  gemm_wmma_kernel<32><<<Ntot / 64, 256, 0, stream>>>(bufB, wp[0], bufA);
  gat_aggregate_kernel<<<Bg, 256, 0, stream>>>(bufA, We[0], As[0], Ad[0], Ae[0], Bb[0],
                                               esrc, edst, ea, eal, 0, bufB);
  // ---- layer 2 ----
  gemm_wmma_kernel<HC><<<Ntot / 64, 256, 0, stream>>>(bufB, wp[1], bufA);
  gat_aggregate_kernel<<<Bg, 256, 0, stream>>>(bufA, We[1], As[1], Ad[1], Ae[1], Bb[1],
                                               esrc, edst, ea, eal, 1, bufB);
  // ---- layer 3 ----
  gemm_wmma_kernel<HC><<<Ntot / 64, 256, 0, stream>>>(bufB, wp[2], bufA);
  gat_aggregate_kernel<<<Bg, 256, 0, stream>>>(bufA, We[2], As[2], Ad[2], Ae[2], Bb[2],
                                               esrc, edst, ea, eal, 1, bufB);

  // ---- head: pool + fc1 + fc2 (reads f16 layer-3 output) ----
  gat_head_kernel<<<Bg, 256, 0, stream>>>(bufB, fc1w, fc1b, fc2w, fc2b, out);
}
